// LSTMModel_71313636982822
// MI455X (gfx1250) — matrix operand order repaired and checked by reading, not compiled, not run on hardware
//
#include <hip/hip_runtime.h>

// ---------------------------------------------------------------------------
// LSTM image-captioning forward for MI455X (gfx1250, wave32, WMMA).
// B=32 S=128 L=2 H=D=512 V=32000 F=768
//
// Structure:
//   XU   = x_embed @ Uh0                  (parallel precompute, f32 WMMA)
//   UW1  = Wxh0 @ Uh1                     (parallel precompute, f32 WMMA)
//   beff1 = bw1 + bu1 + bxh0 @ Uh1        (bias fold)
//   per step: g0 = h0@Wh0 + XU[t] + beff0         -> h0', c0
//             g1 = h1@Wh1 + h0'@UW1 + beff1       -> h1', c1 ; Hist[t] = h1'
//   ys   = Hist @ Wxh1 + bxh1  (one deferred GEMM, f16 out)
//   out  = ys @ Wc + bc        (f16 WMMA, fp32 accum)
// ---------------------------------------------------------------------------

typedef float    v2f  __attribute__((ext_vector_type(2)));
typedef float    v8f  __attribute__((ext_vector_type(8)));
typedef _Float16 v8h  __attribute__((ext_vector_type(8)));
typedef _Float16 v16h __attribute__((ext_vector_type(16)));

#define B_  32
#define S_  128
#define L_  2
#define H_  512
#define D_  512
#define V_  32000
#define F_  768
#define G4  (4 * H_)   // 2048

// fp32 WMMA 16x16x4 (ISA 7.12.2 layouts).
__device__ __forceinline__ v8f wmma4_f32(v2f a, v2f b, v8f c) {
  return __builtin_amdgcn_wmma_f32_16x16x4_f32(false, a, false, b, (short)0, c,
                                               false, false);
}

// ---------------------------------------------------------------------------
// General fp32 GEMM: out[M,N] = A[M,K] @ Bm[K,N] + bias.
// grid = (N/64, M/16), 4 waves/block, one 16x16 tile per wave.
// half_out != 0 -> write _Float16, else float.
// ---------------------------------------------------------------------------
__global__ __launch_bounds__(128) void gemm_f32_kernel(
    const float* __restrict__ A, int lda, int K,
    const float* __restrict__ Bm, int ldb,
    const float* __restrict__ bias,
    void* __restrict__ out, long ostride, int half_out)
{
  const int tid  = threadIdx.x;
  const int wave = tid >> 5;
  const int lane = tid & 31;
  const int lm   = lane & 15;
  const int half = lane >> 4;

  const int m0 = blockIdx.y * 16;
  const int n0 = (blockIdx.x * 4 + wave) * 16;

  const float* Ap = A + (long)m0 * lda;
  const float* Bp = Bm + n0;

  v8f acc = {};
#pragma unroll 4
  for (int k0 = 0; k0 < K; k0 += 4) {
    const int ka = k0 + 2 * half;
    v2f a, b;
    a.x = Ap[lm * lda + ka];
    a.y = Ap[lm * lda + ka + 1];
    b.x = Bp[(long)ka * ldb + lm];
    b.y = Bp[(long)(ka + 1) * ldb + lm];
    acc = wmma4_f32(a, b, acc);
  }

  const float bv = bias ? bias[n0 + lm] : 0.0f;
#pragma unroll
  for (int r = 0; r < 8; ++r) {
    const int row = m0 + r + 8 * half;
    const long idx = (long)row * ostride + n0 + lm;
    const float v = acc[r] + bv;
    if (half_out) ((_Float16*)out)[idx] = (_Float16)v;
    else          ((float*)out)[idx]    = v;
  }
}

// ---------------------------------------------------------------------------
// Fused LSTM gate kernel, one layer, one step:
//   g = h_prev @ Wh  [+ x @ U]  [+ xadd row]  + beff
//   f,i,o = sigmoid; c' = f*c + i*tanh(cc); h' = o*tanh(c')
// One block per 16-column slice of H; 8 waves = 2 row-tiles x 4 gates.
// c updated in place (block-private columns); h double-buffered by caller.
// Optional extra store of h' into Hist (for the deferred projection GEMM).
// ---------------------------------------------------------------------------
__global__ __launch_bounds__(256) void lstm_gates_kernel(
    const float* __restrict__ hprev,  // [32, 512]
    const float* __restrict__ Wh,     // [512, 2048]
    const float* __restrict__ x,      // nullable; rows at stride xstride
    long xstride,
    const float* __restrict__ U,      // [512, 2048] used with x
    const float* __restrict__ xadd,   // nullable; [.,2048] rows at xastride
    long xastride,
    const float* __restrict__ beff,   // [2048] combined bias
    float* __restrict__ c,            // [32, 512] in/out
    float* __restrict__ hout,         // [32, 512]
    float* __restrict__ hist,         // nullable; rows at stride hstride
    long hstride)
{
  __shared__ float gsm[4][32][16];

  const int tid  = threadIdx.x;
  const int wave = tid >> 5;
  const int lane = tid & 31;
  const int lm   = lane & 15;
  const int half = lane >> 4;
  const int mt   = wave & 1;   // row tile (batch 0-15 / 16-31)
  const int ga   = wave >> 1;  // gate: 0=f 1=i 2=o 3=cc
  const int m0   = mt * 16;
  const int n0   = blockIdx.x * 16;   // column within H
  const int col0 = ga * H_ + n0;      // column within 4H

  const float* Ah = hprev + (long)m0 * H_;
  const float* Bh = Wh + col0;

  v8f acc = {};
#pragma unroll 4
  for (int k0 = 0; k0 < H_; k0 += 4) {
    const int ka = k0 + 2 * half;
    v2f a, b;
    a.x = Ah[lm * H_ + ka];
    a.y = Ah[lm * H_ + ka + 1];
    b.x = Bh[(long)ka * G4 + lm];
    b.y = Bh[(long)(ka + 1) * G4 + lm];
    acc = wmma4_f32(a, b, acc);
  }
  if (x) {
    const float* Ax = x + (long)m0 * xstride;
    const float* Bu = U + col0;
#pragma unroll 4
    for (int k0 = 0; k0 < H_; k0 += 4) {
      const int ka = k0 + 2 * half;
      v2f a, b;
      a.x = Ax[(long)lm * xstride + ka];
      a.y = Ax[(long)lm * xstride + ka + 1];
      b.x = Bu[(long)ka * G4 + lm];
      b.y = Bu[(long)(ka + 1) * G4 + lm];
      acc = wmma4_f32(a, b, acc);
    }
  }

#pragma unroll
  for (int r = 0; r < 8; ++r) {
    gsm[ga][m0 + r + 8 * half][lm] = acc[r];
  }
  __syncthreads();

  for (int e = tid; e < 512; e += 256) {
    const int row  = e >> 4;   // batch
    const int cl   = e & 15;
    const int hcol = n0 + cl;
    float f  = gsm[0][row][cl] + beff[0 * H_ + hcol];
    float ig = gsm[1][row][cl] + beff[1 * H_ + hcol];
    float o  = gsm[2][row][cl] + beff[2 * H_ + hcol];
    float cc = gsm[3][row][cl] + beff[3 * H_ + hcol];
    if (xadd) {
      const float* xr = xadd + (long)row * xastride;
      f  += xr[0 * H_ + hcol];
      ig += xr[1 * H_ + hcol];
      o  += xr[2 * H_ + hcol];
      cc += xr[3 * H_ + hcol];
    }
    f  = 1.0f / (1.0f + expf(-f));
    ig = 1.0f / (1.0f + expf(-ig));
    o  = 1.0f / (1.0f + expf(-o));
    const float cn = f * c[row * H_ + hcol] + ig * tanhf(cc);
    const float hv = o * tanhf(cn);
    c[row * H_ + hcol]    = cn;
    hout[row * H_ + hcol] = hv;
    if (hist) hist[(long)row * hstride + hcol] = hv;
  }
}

// ---------------------------------------------------------------------------
// Classifier: out[4096,32000] = A_f16[4096,512] @ Wc + bc; f16 WMMA, fp32 acc.
// 4 waves/block, 4 N-tiles per wave sharing one A fragment (4x A reuse).
// grid = (2000/16, 256). B supplied K-major (WcT[32000,512]).
// ---------------------------------------------------------------------------
__global__ __launch_bounds__(128) void classifier_f16_kernel(
    const _Float16* __restrict__ A,    // [4096, 512]
    const _Float16* __restrict__ Bt,   // [32000, 512] (Wc transposed)
    const float* __restrict__ bias,    // [32000]
    float* __restrict__ out)           // [4096, 32000]
{
  const int tid  = threadIdx.x;
  const int wave = tid >> 5;
  const int lane = tid & 31;
  const int lm   = lane & 15;
  const int half = lane >> 4;

  const int m0    = blockIdx.y * 16;
  const int nbase = (blockIdx.x * 4 + wave) * 64;  // 4 tiles of 16

  const _Float16* Ap = A + (long)m0 * D_ + lm * D_;

  v8f acc0 = {}, acc1 = {}, acc2 = {}, acc3 = {};
#pragma unroll
  for (int k0 = 0; k0 < D_; k0 += 32) {
    // A fragment (16-bit A 16x32 layout): lanes0-15 K {0..7,16..23},
    // lanes16-31 K {8..15,24..31}
    const _Float16* ar = Ap + k0 + 8 * half;
    const v8h a0 = *(const v8h*)(ar);
    const v8h a1 = *(const v8h*)(ar + 16);
    const v16h a = __builtin_shufflevector(a0, a1, 0, 1, 2, 3, 4, 5, 6, 7, 8,
                                           9, 10, 11, 12, 13, 14, 15);
#pragma unroll
    for (int s = 0; s < 4; ++s) {
      // B 32x16: lanes0-15 K {0..7,16..23}, lanes16-31 K {8..15,24..31}
      const _Float16* br =
          Bt + (long)(nbase + 16 * s + lm) * D_ + k0 + 8 * half;
      __builtin_prefetch((const void*)(br + 64), 0, 1);
      const v8h b0 = *(const v8h*)(br);
      const v8h b1 = *(const v8h*)(br + 16);
      const v16h b = __builtin_shufflevector(b0, b1, 0, 1, 2, 3, 4, 5, 6, 7,
                                             8, 9, 10, 11, 12, 13, 14, 15);
      v8f* accp = (s == 0) ? &acc0 : (s == 1) ? &acc1 : (s == 2) ? &acc2 : &acc3;
      *accp = __builtin_amdgcn_wmma_f32_16x16x32_f16(false, a, false, b,
                                                     (short)0, *accp, false,
                                                     false);
    }
  }

#pragma unroll
  for (int s = 0; s < 4; ++s) {
    const v8f acc = (s == 0) ? acc0 : (s == 1) ? acc1 : (s == 2) ? acc2 : acc3;
    const int n0 = nbase + 16 * s;
    const float bv = bias[n0 + lm];
#pragma unroll
    for (int r = 0; r < 8; ++r) {
      const int row = m0 + r + 8 * half;
      out[(long)row * V_ + n0 + lm] = acc[r] + bv;
    }
  }
}

// ---------------------------------------------------------------------------
// Small utility kernels
// ---------------------------------------------------------------------------
__global__ void gather_embed_kernel(const int* __restrict__ tokens,
                                    const float* __restrict__ embed,
                                    float* __restrict__ xe) {
  const int r = blockIdx.x;  // b*S + s
  const int t = tokens[r];
  const float4* src = (const float4*)(embed + (long)t * D_);
  float4* dst = (float4*)(xe + (long)r * D_);
  dst[threadIdx.x] = src[threadIdx.x];  // 128 threads x float4 = 512 floats
}

// LDS-tiled transpose+convert: Wc[512,32000] f32 -> WcT[32000,512] f16.
// 64x64 tiles; coalesced reads along V, coalesced f16 writes along D.
// grid = (V/64, D/64), 256 threads.
__global__ __launch_bounds__(256) void transpose_f16_kernel(
    const float* __restrict__ Wc, _Float16* __restrict__ WcT) {
  __shared__ _Float16 tile[64][65];   // padded: no bank conflicts
  const int n0 = blockIdx.x * 64;     // V offset
  const int k0 = blockIdx.y * 64;     // D offset
  const int tid = threadIdx.x;

#pragma unroll
  for (int e = 0; e < 16; ++e) {
    const int idx = tid + e * 256;    // 0..4095
    const int r = idx >> 6;           // k within tile
    const int cl = idx & 63;          // n within tile
    tile[r][cl] = (_Float16)Wc[(long)(k0 + r) * V_ + n0 + cl];
  }
  __syncthreads();
#pragma unroll
  for (int e = 0; e < 16; ++e) {
    const int idx = tid + e * 256;
    const int r = idx >> 6;           // n within tile
    const int cl = idx & 63;          // k within tile
    WcT[(long)(n0 + r) * D_ + k0 + cl] = tile[cl][r];
  }
}

// beff[0][n] = bw0+bu0 ; beff[1][n] = bw1+bu1 + (bxh0 @ Uh1)[n]
__global__ void bias_eff_kernel(const float* __restrict__ bw,
                                const float* __restrict__ bu,
                                const float* __restrict__ bxh0,
                                const float* __restrict__ Uh1,  // [512,2048]
                                float* __restrict__ beff) {     // [2,2048]
  const int n = blockIdx.x * blockDim.x + threadIdx.x;
  if (n >= G4) return;
  beff[n] = bw[n] + bu[n];
  float s = 0.0f;
  for (int k = 0; k < D_; ++k) s += bxh0[k] * Uh1[(long)k * G4 + n];
  beff[G4 + n] = bw[G4 + n] + bu[G4 + n] + s;
}

// ---------------------------------------------------------------------------
// Host launcher
// ---------------------------------------------------------------------------
extern "C" void kernel_launch(void* const* d_in, const int* in_sizes, int n_in,
                              void* d_out, int out_size, void* d_ws,
                              size_t ws_size, hipStream_t stream) {
  const float* im_feat = (const float*)d_in[0];
  const int*   tokens  = (const int*)d_in[1];
  const float* h0      = (const float*)d_in[2];
  const float* c0      = (const float*)d_in[3];
  const float* embed   = (const float*)d_in[4];
  const float* W_im    = (const float*)d_in[5];
  const float* b_im    = (const float*)d_in[6];
  const float* Wh      = (const float*)d_in[7];
  const float* bw      = (const float*)d_in[8];
  const float* Uh      = (const float*)d_in[9];
  const float* bu      = (const float*)d_in[10];
  const float* Wxh     = (const float*)d_in[11];
  const float* bxh     = (const float*)d_in[12];
  const float* Wc      = (const float*)d_in[13];
  const float* bc      = (const float*)d_in[14];
  float* out = (float*)d_out;

  char* ws = (char*)d_ws;
  size_t off = 0;
  auto alloc = [&](size_t bytes) -> void* {
    void* p = ws + off;
    off = (off + bytes + 255) & ~(size_t)255;
    return p;
  };
  float*    xe   = (float*)alloc((size_t)B_ * S_ * D_ * 4);   // embedded tokens
  float*    XU   = (float*)alloc((size_t)B_ * S_ * G4 * 4);   // x_embed @ Uh0
  float*    XUim = (float*)alloc((size_t)B_ * G4 * 4);        // y_im @ Uh0
  float*    UW1  = (float*)alloc((size_t)D_ * G4 * 4);        // Wxh0 @ Uh1
  float*    Hist = (float*)alloc((size_t)B_ * S_ * H_ * 4);   // h1' history
  float*    y_im = (float*)alloc((size_t)B_ * D_ * 4);
  float*    hbuf = (float*)alloc((size_t)2 * L_ * B_ * H_ * 4);
  float*    cbuf = (float*)alloc((size_t)L_ * B_ * H_ * 4);
  float*    beff = (float*)alloc((size_t)2 * G4 * 4);
  _Float16* ysh  = (_Float16*)alloc((size_t)B_ * S_ * D_ * 2);
  _Float16* WcT  = (_Float16*)alloc((size_t)D_ * V_ * 2);

  const size_t BH = (size_t)B_ * H_;

  hipMemcpyAsync(hbuf, h0, (size_t)L_ * B_ * H_ * 4,
                 hipMemcpyDeviceToDevice, stream);
  hipMemcpyAsync(cbuf, c0, (size_t)L_ * B_ * H_ * 4,
                 hipMemcpyDeviceToDevice, stream);

  // Wc -> transposed f16 (consumed only by the final classifier)
  transpose_f16_kernel<<<dim3(V_ / 64, D_ / 64), 256, 0, stream>>>(Wc, WcT);
  gather_embed_kernel<<<dim3(B_ * S_), 128, 0, stream>>>(tokens, embed, xe);

  // y_im = im_feat @ W_im + b_im            [32,768]x[768,512]
  gemm_f32_kernel<<<dim3(D_ / 64, B_ / 16), 128, 0, stream>>>(
      im_feat, F_, F_, W_im, D_, b_im, y_im, (long)D_, 0);
  // XUim = y_im @ Uh0                       [32,512]x[512,2048]
  gemm_f32_kernel<<<dim3(G4 / 64, B_ / 16), 128, 0, stream>>>(
      y_im, D_, D_, Uh, G4, nullptr, XUim, (long)G4, 0);
  // XU = x_embed @ Uh0                      [4096,512]x[512,2048]
  gemm_f32_kernel<<<dim3(G4 / 64, (B_ * S_) / 16), 128, 0, stream>>>(
      xe, D_, D_, Uh, G4, nullptr, XU, (long)G4, 0);
  // UW1 = Wxh0 @ Uh1                        [512,512]x[512,2048]
  gemm_f32_kernel<<<dim3(G4 / 64, H_ / 16), 128, 0, stream>>>(
      Wxh, D_, H_, Uh + (size_t)D_ * G4, G4, nullptr, UW1, (long)G4, 0);
  // combined biases (incl. bxh0 folded through Uh1)
  bias_eff_kernel<<<dim3(G4 / 256), 256, 0, stream>>>(
      bw, bu, bxh, Uh + (size_t)D_ * G4, beff);

  // ---- sequential recurrence: 2 kernels per step --------------------------
  int cur = 0;
  for (int t = -1; t < S_; ++t) {
    float* hc = hbuf + (size_t)cur * L_ * B_ * H_;
    float* hn = hbuf + (size_t)(cur ^ 1) * L_ * B_ * H_;
    const float* xa;
    long xas;
    if (t < 0) { xa = XUim;                xas = G4; }             // warm-up
    else       { xa = XU + (long)t * G4;   xas = (long)S_ * G4; }  // token t

    // layer 0: g0 = h0@Wh0 + XU-row + beff0
    lstm_gates_kernel<<<32, 256, 0, stream>>>(
        hc, Wh, nullptr, 0, nullptr, xa, xas, beff, cbuf, hn, nullptr, 0);

    // layer 1: g1 = h1@Wh1 + h0'@UW1 + beff1 ; Hist[t] = h1'
    float* histp = (t >= 0) ? (Hist + (long)t * H_) : nullptr;
    lstm_gates_kernel<<<32, 256, 0, stream>>>(
        hc + BH, Wh + (size_t)H_ * G4, hn, (long)H_, UW1,
        nullptr, 0, beff + G4, cbuf + BH, hn + BH, histp, (long)S_ * H_);

    cur ^= 1;
  }

  // deferred top-layer projection, f16 output: ysh = Hist @ Wxh1 + bxh1
  gemm_f32_kernel<<<dim3(D_ / 64, (B_ * S_) / 16), 128, 0, stream>>>(
      Hist, H_, H_, Wxh + (size_t)H_ * D_, D_, bxh + D_, ysh, (long)D_, 1);

  // classifier: logits = ysh @ Wc + bc (f16 WMMA, fp32 accum)
  classifier_f16_kernel<<<dim3((V_ / 16) / 16, (B_ * S_) / 16), 128, 0,
                          stream>>>(ysh, WcT, bc, out);
}
